// LoadBalancedMoE_19868518711312
// MI455X (gfx1250) — matrix operand order repaired and checked by reading, not compiled, not run on hardware
//
#include <hip/hip_runtime.h>
#include <math.h>

// ---------------------------------------------------------------- types
typedef __bf16 bf16;
typedef __attribute__((ext_vector_type(16))) __bf16 v16bf;
typedef __attribute__((ext_vector_type(8)))  float  v8f;

// ---------------------------------------------------------------- dims
#define Hc   1024
#define NHc  4
#define DHc  256
#define Ec   8
#define Ic   4096
#define I2c  2048
#define Bc   2
#define Sc   1024
#define Tc   (Bc * Sc)   // 2048 tokens

#define BM 128
#define BN 128
#define BK 32

// ---------------------------------------------------------------- helpers
static __device__ inline bf16 f2bf(float f) {
  union { float f; unsigned u; } a; a.f = f;
  unsigned r = a.u + 0x7fffu + ((a.u >> 16) & 1u);   // round-to-nearest-even
  unsigned short h = (unsigned short)(r >> 16);
  bf16 b; __builtin_memcpy(&b, &h, 2); return b;
}

static __device__ inline float gelu_exact(float x) {
  return 0.5f * x * (1.0f + erff(x * 0.70710678118654752440f));
}

// ---------------------------------------------------------------- elementwise
__global__ __launch_bounds__(256) void zero_f32_kernel(float* p, size_t n) {
  size_t i = (size_t)blockIdx.x * blockDim.x + threadIdx.x;
  size_t stride = (size_t)gridDim.x * blockDim.x;
  for (; i < n; i += stride) p[i] = 0.0f;
}

__global__ __launch_bounds__(256) void cast_f32_bf16_kernel(const float* __restrict__ in,
                                                            bf16* __restrict__ out, size_t n) {
  size_t i = (size_t)blockIdx.x * blockDim.x + threadIdx.x;
  size_t stride = (size_t)gridDim.x * blockDim.x;
  for (; i < n; i += stride) out[i] = f2bf(in[i]);
}

// ---------------------------------------------------------------- WMMA GEMM
// C[M,N] = act(A[M,K] @ B[K,N] + bias), A/B bf16 row-major, C -> f32 and/or bf16.
// act: 0=none, 1=gelu(exact), 2=relu.  M%128==0, N%128==0, K%32==0.
// Block tile 128x128, double-buffered LDS, 8 waves, each wave 32Mx64N (8 WMMA/K-step).
__global__ __launch_bounds__(256)
void gemm_bf16_wmma_kernel(const bf16* __restrict__ A, const bf16* __restrict__ B,
                           const float* __restrict__ bias,
                           float* __restrict__ Cf, bf16* __restrict__ Cb,
                           int M, int N, int Kd, int act) {
  __shared__ __align__(16) bf16 As[2][BM][BK];   // 2 x 8 KB
  __shared__ __align__(16) bf16 Bs[2][BN][BK];   // 2 x 8 KB, stored [n][k] (transposed)

  const int tid  = threadIdx.x;
  const int lane = tid & 31;
  const int wave = tid >> 5;            // 8 waves: 4 (M) x 2 (N)
  const int wm   = (wave & 3) * 32;     // wave M offset inside block tile
  const int wn   = (wave >> 2) * 64;    // wave N offset inside block tile
  const int bm   = blockIdx.y * BM;
  const int bn   = blockIdx.x * BN;

  const v8f zero8 = {0.f, 0.f, 0.f, 0.f, 0.f, 0.f, 0.f, 0.f};
  v8f acc[2][4];
#pragma unroll
  for (int mi = 0; mi < 2; ++mi)
#pragma unroll
    for (int ni = 0; ni < 4; ++ni) acc[mi][ni] = zero8;

  const int rlo   = lane & 15;           // row (A) / col (B) within 16
  const int khalf = (lane >> 4) * 8;     // A: K base of elems 0..7 (0 or 8)
  const int kb    = (lane >> 4) * 8;     // B: K base of elems 0..7 (0 or 8)

  // per-thread staging coordinates (two b128 chunks each for A and B)
  int ar[2], ac[2], bkk[2], bnn[2];
#pragma unroll
  for (int i = 0; i < 2; ++i) {
    int aidx = (tid + i * 256) * 8;      // A elements 0..4095 (128x32)
    ar[i] = aidx >> 5;                   // row
    ac[i] = aidx & 31;                   // k
    int bidx = (tid + i * 256) * 8;      // B elements 0..4095 (32x128)
    bkk[i] = bidx >> 7;                  // k row 0..31
    bnn[i] = bidx & 127;                 // n start (multiple of 8)
  }

  // ---- prologue: stage 0
  uint4 areg[2], breg[2];
#pragma unroll
  for (int i = 0; i < 2; ++i) {
    areg[i] = *(const uint4*)&A[(size_t)(bm + ar[i]) * Kd + ac[i]];
    breg[i] = *(const uint4*)&B[(size_t)bkk[i] * N + (bn + bnn[i])];
  }
#pragma unroll
  for (int i = 0; i < 2; ++i) {
    *(uint4*)&As[0][ar[i]][ac[i]] = areg[i];
    const bf16* tp = reinterpret_cast<const bf16*>(&breg[i]);
#pragma unroll
    for (int j = 0; j < 8; ++j) Bs[0][bnn[i] + j][bkk[i]] = tp[j];
  }
  __syncthreads();

  for (int k0 = 0; k0 < Kd; k0 += BK) {
    const int cur = (k0 >> 5) & 1;
    const int nxt = cur ^ 1;
    const bool more = (k0 + BK) < Kd;

    // ---- issue next tile's global loads early (land them during WMMA burst)
    if (more) {
#pragma unroll
      for (int i = 0; i < 2; ++i) {
        areg[i] = *(const uint4*)&A[(size_t)(bm + ar[i]) * Kd + (k0 + BK + ac[i])];
        breg[i] = *(const uint4*)&B[(size_t)(k0 + BK + bkk[i]) * N + (bn + bnn[i])];
      }
      if (k0 + 2 * BK < Kd)
        __builtin_prefetch(&A[(size_t)(bm + ar[0]) * Kd + (k0 + 2 * BK + ac[0])], 0, 1);
    }

    // ---- fragments per CDNA5 VGPR layouts
    v16bf afrag[2], bfrag[4];
#pragma unroll
    for (int mi = 0; mi < 2; ++mi) {
      int r = wm + mi * 16 + rlo;
      union { v16bf v; uint4 u[2]; } ua;
      ua.u[0] = *(const uint4*)&As[cur][r][khalf];        // K = khalf..khalf+7
      ua.u[1] = *(const uint4*)&As[cur][r][khalf + 16];   // K = khalf+16..+23
      afrag[mi] = ua.v;
    }
#pragma unroll
    for (int ni = 0; ni < 4; ++ni) {
      int c = wn + ni * 16 + rlo;
      union { v16bf v; uint4 u[2]; } ub;
      ub.u[0] = *(const uint4*)&Bs[cur][c][kb];           // K = kb..kb+7
      ub.u[1] = *(const uint4*)&Bs[cur][c][kb + 16];      // K = kb+16..kb+23
      bfrag[ni] = ub.v;
    }
#pragma unroll
    for (int mi = 0; mi < 2; ++mi)
#pragma unroll
      for (int ni = 0; ni < 4; ++ni)
        acc[mi][ni] = __builtin_amdgcn_wmma_f32_16x16x32_bf16(
            false, afrag[mi], false, bfrag[ni], (short)0, acc[mi][ni],
            false, false);

    // ---- stage next tile into the alternate LDS buffer
    if (more) {
#pragma unroll
      for (int i = 0; i < 2; ++i) {
        *(uint4*)&As[nxt][ar[i]][ac[i]] = areg[i];
        const bf16* tp = reinterpret_cast<const bf16*>(&breg[i]);
#pragma unroll
        for (int j = 0; j < 8; ++j) Bs[nxt][bnn[i] + j][bkk[i]] = tp[j];
      }
    }
    __syncthreads();   // readers of `cur` done before next iter overwrites it
  }

  // --- epilogue: C layout — vgpr j: M = j (+8 for hi lanes), N = lane&15
  const int rhi = (lane >> 4) * 8;
#pragma unroll
  for (int mi = 0; mi < 2; ++mi) {
#pragma unroll
    for (int ni = 0; ni < 4; ++ni) {
      int col = bn + wn + ni * 16 + rlo;
      float bv = bias ? bias[col] : 0.0f;
#pragma unroll
      for (int j = 0; j < 8; ++j) {
        int row = bm + wm + mi * 16 + rhi + j;
        float v = acc[mi][ni][j] + bv;
        if (act == 1)      v = gelu_exact(v);
        else if (act == 2) v = fmaxf(v, 0.0f);
        size_t off = (size_t)row * N + col;
        if (Cf) Cf[off] = v;
        if (Cb) Cb[off] = f2bf(v);
      }
    }
  }
}

// ---------------------------------------------------------------- attention
// One block per (b, h, q): softmax(Q.K^T/sqrt(DH)) @ V  -> O[T,H]
__global__ __launch_bounds__(256)
void attn_kernel(const float* __restrict__ Q, const float* __restrict__ Kt,
                 const float* __restrict__ V, float* __restrict__ O) {
  __shared__ float sc[Sc];
  __shared__ float qs[DHc];
  __shared__ float red[256];
  const int tid = threadIdx.x;
  const int q = blockIdx.x, h = blockIdx.y, b = blockIdx.z;
  const size_t rowq = (size_t)(b * Sc + q) * Hc + h * DHc;

  for (int d = tid; d < DHc; d += 256) qs[d] = Q[rowq + d];
  __syncthreads();

  const float scale = 0.0625f;  // 1/sqrt(256)
  for (int k = tid; k < Sc; k += 256) {
    const float* kr = Kt + (size_t)(b * Sc + k) * Hc + h * DHc;
    float dot = 0.0f;
    for (int d = 0; d < DHc; ++d) dot += qs[d] * kr[d];
    sc[k] = dot * scale;
  }
  __syncthreads();

  float m = -INFINITY;
  for (int k = tid; k < Sc; k += 256) m = fmaxf(m, sc[k]);
  red[tid] = m; __syncthreads();
  for (int s = 128; s > 0; s >>= 1) {
    if (tid < s) red[tid] = fmaxf(red[tid], red[tid + s]);
    __syncthreads();
  }
  float mx = red[0]; __syncthreads();

  float ps = 0.0f;
  for (int k = tid; k < Sc; k += 256) { float e = __expf(sc[k] - mx); sc[k] = e; ps += e; }
  red[tid] = ps; __syncthreads();
  for (int s = 128; s > 0; s >>= 1) {
    if (tid < s) red[tid] += red[tid + s];
    __syncthreads();
  }
  float inv = 1.0f / red[0];
  __syncthreads();

  // DHc == blockDim: thread = output channel
  const float* vb = V + (size_t)(b * Sc) * Hc + h * DHc + tid;
  float o = 0.0f;
  for (int k = 0; k < Sc; ++k) o += sc[k] * vb[(size_t)k * Hc];
  O[rowq + tid] = o * inv;
}

// ---------------------------------------------------------------- router
__global__ __launch_bounds__(256)
void router_logits_kernel(const float* __restrict__ r1, const float* __restrict__ Wr2,
                          const float* __restrict__ br2, float* __restrict__ logits) {
  int idx = blockIdx.x * blockDim.x + threadIdx.x;   // t*E + e
  if (idx >= Tc * Ec) return;
  int t = idx >> 3, e = idx & 7;
  const float* r = r1 + (size_t)t * (Hc / 2);
  float s = br2[e];
  for (int k = 0; k < Hc / 2; ++k) s += r[k] * Wr2[k * Ec + e];
  logits[idx] = s;
}

__global__ __launch_bounds__(256)
void router_topk_kernel(const float* __restrict__ logits, const float* __restrict__ escale,
                        float* __restrict__ combine, float* __restrict__ usage) {
  int t = blockIdx.x * blockDim.x + threadIdx.x;
  if (t >= Tc) return;
  float l[Ec];
  float mx = -INFINITY;
  for (int e = 0; e < Ec; ++e) { l[e] = logits[(size_t)t * Ec + e]; mx = fmaxf(mx, l[e]); }
  float sum = 0.0f;
  for (int e = 0; e < Ec; ++e) { l[e] = __expf(l[e] - mx); sum += l[e]; }
  float inv = 1.0f / sum;
  for (int e = 0; e < Ec; ++e) l[e] *= inv;          // probs
  int i0 = 0, i1 = -1;
  for (int e = 1; e < Ec; ++e) if (l[e] > l[i0]) i0 = e;   // ties -> lowest idx
  for (int e = 0; e < Ec; ++e) {
    if (e == i0) continue;
    if (i1 < 0 || l[e] > l[i1]) i1 = e;
  }
  // softmax over the two selected probabilities (as in reference)
  float a = l[i0], b2 = l[i1];
  float e0 = __expf(a - a), e1 = __expf(b2 - a);
  float z = 1.0f / (e0 + e1);
  float w0 = e0 * z, w1 = e1 * z;
  for (int e = 0; e < Ec; ++e) {
    float w = (e == i0) ? w0 : (e == i1) ? w1 : 0.0f;
    combine[(size_t)t * Ec + e] = w * escale[e];
  }
  atomicAdd(&usage[i0], 1.0f);
  atomicAdd(&usage[i1], 1.0f);
}

__global__ void lb_loss_kernel(const float* __restrict__ usage, float* __restrict__ loss) {
  if (threadIdx.x == 0 && blockIdx.x == 0) {
    float s = 0.0f;
    for (int e = 0; e < Ec; ++e) s += usage[e];
    float mean = s / Ec;
    float v = 0.0f;
    for (int e = 0; e < Ec; ++e) { float d = usage[e] - mean; v += d * d; }
    float stdv = sqrtf(v / (Ec - 1));                 // ddof=1
    *loss = stdv / (mean + 1e-6f) * 0.01f;
  }
}

// ---------------------------------------------------------------- LN + combine
__global__ __launch_bounds__(256)
void ln_combine_kernel(const float* __restrict__ h3, const float* __restrict__ g,
                       const float* __restrict__ bta, const float* __restrict__ combine,
                       int e, float* __restrict__ out) {
  __shared__ float red[256];
  const int t = blockIdx.x, tid = threadIdx.x;
  const float* x = h3 + (size_t)t * Hc;
  float s = 0.0f;
  for (int i = tid; i < Hc; i += 256) s += x[i];
  red[tid] = s; __syncthreads();
  for (int st = 128; st > 0; st >>= 1) { if (tid < st) red[tid] += red[tid + st]; __syncthreads(); }
  float mu = red[0] * (1.0f / Hc); __syncthreads();
  float v = 0.0f;
  for (int i = tid; i < Hc; i += 256) { float d = x[i] - mu; v += d * d; }
  red[tid] = v; __syncthreads();
  for (int st = 128; st > 0; st >>= 1) { if (tid < st) red[tid] += red[tid + st]; __syncthreads(); }
  float rs = rsqrtf(red[0] * (1.0f / Hc) + 1e-5f);
  float cw = combine[(size_t)t * Ec + e];
  for (int i = tid; i < Hc; i += 256) {
    float val = (x[i] - mu) * rs * g[i] + bta[i];
    out[(size_t)t * Hc + i] += cw * val;
  }
}

// ---------------------------------------------------------------- host side
static void launch_cast(const float* in, bf16* out, size_t n, hipStream_t s) {
  int blocks = (int)((n + 256 * 8 - 1) / (256 * 8));
  if (blocks > 4096) blocks = 4096;
  if (blocks < 1) blocks = 1;
  cast_f32_bf16_kernel<<<blocks, 256, 0, s>>>(in, out, n);
}

static void launch_gemm(const bf16* A, const bf16* B, const float* bias,
                        float* Cf, bf16* Cb, int M, int N, int K, int act,
                        hipStream_t s) {
  dim3 grid(N / BN, M / BM);
  gemm_bf16_wmma_kernel<<<grid, 256, 0, s>>>(A, B, bias, Cf, Cb, M, N, K, act);
}

extern "C" void kernel_launch(void* const* d_in, const int* in_sizes, int n_in,
                              void* d_out, int out_size, void* d_ws, size_t ws_size,
                              hipStream_t stream) {
  (void)in_sizes; (void)n_in; (void)out_size; (void)ws_size;
  // inputs, setup_inputs() order
  const float* x    = (const float*)d_in[0];
  const float* Wq   = (const float*)d_in[1];  const float* bq  = (const float*)d_in[2];
  const float* Wk   = (const float*)d_in[3];  const float* bk  = (const float*)d_in[4];
  const float* Wv   = (const float*)d_in[5];  const float* bv  = (const float*)d_in[6];
  const float* Wo   = (const float*)d_in[7];  const float* bo  = (const float*)d_in[8];
  const float* Wr1  = (const float*)d_in[9];  const float* br1 = (const float*)d_in[10];
  const float* Wr2  = (const float*)d_in[11]; const float* br2 = (const float*)d_in[12];
  const float* We1  = (const float*)d_in[13]; const float* be1 = (const float*)d_in[14];
  const float* We2  = (const float*)d_in[15]; const float* be2 = (const float*)d_in[16];
  const float* We3  = (const float*)d_in[17]; const float* be3 = (const float*)d_in[18];
  const float* lng  = (const float*)d_in[19]; const float* lnb = (const float*)d_in[20];
  const float* escl = (const float*)d_in[21];

  float* out = (float*)d_out;                  // [T*H] output + [1] lb_loss

  // workspace carve-up
  char* w = (char*)d_ws;
  auto alloc = [&](size_t bytes) { char* p = w; w += (bytes + 255) & ~(size_t)255; return p; };
  bf16*  xb      = (bf16*) alloc((size_t)Tc * Hc * 2);         // bf16 tokens
  bf16*  wb      = (bf16*) alloc((size_t)Ic * I2c * 2);        // weight staging (16MB)
  bf16*  ab      = (bf16*) alloc((size_t)Tc * Ic * 2);         // bf16 activations
  bf16*  ab2     = (bf16*) alloc((size_t)Tc * I2c * 2);
  float* Qf      = (float*)alloc((size_t)Tc * Hc * 4);
  float* Kf      = (float*)alloc((size_t)Tc * Hc * 4);
  float* Vf      = (float*)alloc((size_t)Tc * Hc * 4);
  float* attnO   = (float*)alloc((size_t)Tc * Hc * 4);
  float* aproj   = (float*)alloc((size_t)Tc * Hc * 4);
  float* r1f     = (float*)alloc((size_t)Tc * (Hc / 2) * 4);
  float* h3f     = (float*)alloc((size_t)Tc * Hc * 4);
  float* logitsf = (float*)alloc((size_t)Tc * Ec * 4);
  float* combf   = (float*)alloc((size_t)Tc * Ec * 4);
  float* usage   = (float*)alloc((size_t)Ec * 4);

  // 0) init accumulators
  zero_f32_kernel<<<1024, 256, 0, stream>>>(out, (size_t)Tc * Hc + 1);
  zero_f32_kernel<<<1, 32, 0, stream>>>(usage, (size_t)Ec);

  // 1) token cast
  launch_cast(x, xb, (size_t)Tc * Hc, stream);

  // 2) Q/K/V projections (WMMA)
  launch_cast(Wq, wb, (size_t)Hc * Hc, stream);
  launch_gemm(xb, wb, bq, Qf, nullptr, Tc, Hc, Hc, 0, stream);
  launch_cast(Wk, wb, (size_t)Hc * Hc, stream);
  launch_gemm(xb, wb, bk, Kf, nullptr, Tc, Hc, Hc, 0, stream);
  launch_cast(Wv, wb, (size_t)Hc * Hc, stream);
  launch_gemm(xb, wb, bv, Vf, nullptr, Tc, Hc, Hc, 0, stream);

  // 3) attention
  attn_kernel<<<dim3(Sc, NHc, Bc), 256, 0, stream>>>(Qf, Kf, Vf, attnO);
  launch_cast(attnO, ab, (size_t)Tc * Hc, stream);

  // 4) output projection
  launch_cast(Wo, wb, (size_t)Hc * Hc, stream);
  launch_gemm(ab, wb, bo, aproj, nullptr, Tc, Hc, Hc, 0, stream);
  launch_cast(aproj, ab, (size_t)Tc * Hc, stream);

  // 5) router MLP layer 1 (ReLU)
  launch_cast(Wr1, wb, (size_t)Hc * (Hc / 2), stream);
  launch_gemm(ab, wb, br1, r1f, nullptr, Tc, Hc / 2, Hc, 2, stream);

  // 6) router logits (N=8, VALU) + top-2 + loss
  router_logits_kernel<<<(Tc * Ec + 255) / 256, 256, 0, stream>>>(r1f, Wr2, br2, logitsf);
  router_topk_kernel<<<(Tc + 255) / 256, 256, 0, stream>>>(logitsf, escl, combf, usage);
  lb_loss_kernel<<<1, 32, 0, stream>>>(usage, out + (size_t)Tc * Hc);

  // 7) dense grouped experts (WMMA) with fused GELU, LN + weighted accumulate
  for (int e = 0; e < Ec; ++e) {
    launch_cast(We1 + (size_t)e * Hc * Ic, wb, (size_t)Hc * Ic, stream);
    launch_gemm(xb, wb, be1 + (size_t)e * Ic, nullptr, ab, Tc, Ic, Hc, 1, stream);

    launch_cast(We2 + (size_t)e * Ic * I2c, wb, (size_t)Ic * I2c, stream);
    launch_gemm(ab, wb, be2 + (size_t)e * I2c, nullptr, ab2, Tc, I2c, Ic, 1, stream);

    launch_cast(We3 + (size_t)e * I2c * Hc, wb, (size_t)I2c * Hc, stream);
    launch_gemm(ab2, wb, be3 + (size_t)e * Hc, h3f, nullptr, Tc, Hc, I2c, 0, stream);

    ln_combine_kernel<<<Tc, 256, 0, stream>>>(h3f, lng + (size_t)e * Hc,
                                              lnb + (size_t)e * Hc, combf, e, out);
  }
}
